// GAT_18056042512920
// MI455X (gfx1250) — matrix operand order repaired and checked by reading, not compiled, not run on hardware
//
#include <hip/hip_runtime.h>

typedef __bf16 bf16_t;
typedef __attribute__((ext_vector_type(16))) __bf16 v16bf;
typedef __attribute__((ext_vector_type(8)))  float  v8f;

#define BATCH 16
#define NN    1024
#define DD    256
#define NEGV  (-1e9f)
#define SCALEV 0.0625f   // 1/sqrt(256)

union frag16 {
    uint4 q[2];
    v16bf v;
};

// ---------------------------------------------------------------------------
// f32 -> bf16 conversion
// ---------------------------------------------------------------------------
__global__ __launch_bounds__(256) void cvt_f32_to_bf16(const float* __restrict__ src,
                                                       bf16_t* __restrict__ dst, int n) {
    int i = blockIdx.x * 256 + threadIdx.x;
    if (i < n) dst[i] = (bf16_t)src[i];
}

// ---------------------------------------------------------------------------
// Y[M,D] = A[M,D] @ Bw[D,D]^T + bias   (all K along contiguous dim)
// optional transposed bf16 copy YT[b][e][n] for use as WMMA B-operand later.
// Block: 256 threads = 8 waves; block tile 128(M) x 128(N); wave tile 16 x 128.
// ---------------------------------------------------------------------------
__global__ __launch_bounds__(256) void gemm_bias_bf16(
    const bf16_t* __restrict__ A,    // [16384, 256] row-major
    const bf16_t* __restrict__ Bw,   // [256, 256] row-major ([e][d]) == B^T
    const float*  __restrict__ bias, // [256]
    bf16_t* __restrict__ Y,          // [16384, 256]
    bf16_t* __restrict__ YT)         // [16][256][1024] or nullptr
{
    const int lane  = threadIdx.x & 31;
    const int wave  = threadIdx.x >> 5;
    const int m0    = blockIdx.x * 128 + wave * 16;
    const int nbase = blockIdx.y * 128;
    const int half  = lane >> 4;
    const int lrow  = lane & 15;   // A row (within tile) / B,C column

    v8f acc[8];
    #pragma unroll
    for (int t = 0; t < 8; ++t) acc[t] = {};

    const uint32_t* arow = (const uint32_t*)(A + (size_t)(m0 + lrow) * DD);
    for (int k0 = 0; k0 < DD; k0 += 32) {
        frag16 af;
        const uint32_t* ap = arow + (k0 >> 1);
        af.q[0] = *(const uint4*)(ap + half * 4);
        af.q[1] = *(const uint4*)(ap + 8 + half * 4);
        #pragma unroll
        for (int t = 0; t < 8; ++t) {
            frag16 bfrag;
            const uint32_t* bp =
                (const uint32_t*)(Bw + (size_t)(nbase + t * 16 + lrow) * DD + k0) + half * 4;
            bfrag.q[0] = *(const uint4*)(bp);
            bfrag.q[1] = *(const uint4*)(bp + 8);
            acc[t] = __builtin_amdgcn_wmma_f32_16x16x32_bf16(
                false, af.v, false, bfrag.v, (short)0, acc[t], false, false);
        }
    }

    // epilogue: +bias, store bf16 row-major (and transposed)
    #pragma unroll
    for (int t = 0; t < 8; ++t) {
        const int col = nbase + t * 16 + lrow;
        const float bv = bias[col];
        union { bf16_t h[8]; uint4 q; } tp;
        #pragma unroll
        for (int r = 0; r < 8; ++r) {
            float v = acc[t][r] + bv;
            int row = m0 + half * 8 + r;
            bf16_t hv = (bf16_t)v;
            Y[(size_t)row * DD + col] = hv;
            tp.h[r] = hv;
        }
        if (YT) {
            const int b    = m0 >> 10;
            const int nloc = (m0 & 1023) + half * 8;     // contiguous over r
            *(uint4*)(YT + ((size_t)b * DD + col) * NN + nloc) = tp.q;
        }
    }
}

// ---------------------------------------------------------------------------
// Fused GAT attention (flash-style, one wave = 16 query rows x 128 out cols).
// S = leakyrelu(scale * q k^T) masked by graph; online softmax; O = relu(P h / l).
// gid layout: [b][mtile][dhalf]; 16*64*2 = 2048 waves.
// ---------------------------------------------------------------------------
__global__ __launch_bounds__(256) void gat_attention(
    const bf16_t* __restrict__ Q,   // [B*N, D] bf16
    const bf16_t* __restrict__ K,   // [B*N, D] bf16
    const bf16_t* __restrict__ HT,  // [B][D][N] bf16 (h transposed)
    const int*    __restrict__ graph, // [B][N][N]
    float*  __restrict__ Of32,      // [B*N, D]
    bf16_t* __restrict__ Obf)       // [B*N, D] (next-layer input)
{
    const int lane  = threadIdx.x & 31;
    const int wave  = threadIdx.x >> 5;
    const int gid   = blockIdx.x * 8 + wave;
    const int dhalf = gid & 1;
    const int mtile = (gid >> 1) & 63;
    const int b     = gid >> 7;
    const int half  = lane >> 4;
    const int lrow  = lane & 15;
    const int qrow0 = b * NN + mtile * 16;

    __shared__ bf16_t pstage[8][16 * 32];
    bf16_t* pl = pstage[wave];

    // resident q fragments: 8 K-chunks of 16x32
    frag16 qf[8];
    {
        const uint32_t* arow = (const uint32_t*)(Q + (size_t)(qrow0 + lrow) * DD);
        #pragma unroll
        for (int kc = 0; kc < 8; ++kc) {
            const uint32_t* ap = arow + kc * 16;
            qf[kc].q[0] = *(const uint4*)(ap + half * 4);
            qf[kc].q[1] = *(const uint4*)(ap + 8 + half * 4);
        }
    }

    v8f o[8];
    #pragma unroll
    for (int t = 0; t < 8; ++t) o[t] = {};
    float mrow[8], lsum[8];
    #pragma unroll
    for (int r = 0; r < 8; ++r) { mrow[r] = -3.0e38f; lsum[r] = 0.f; }

    const int* grow = graph + ((size_t)b * NN + mtile * 16) * NN;

    for (int j = 0; j < 64; j += 2) {
        // ---- S tiles for key columns j*16 .. j*16+31 ----
        v8f s[2];
        s[0] = {}; s[1] = {};
        #pragma unroll
        for (int kc = 0; kc < 8; ++kc) {
            #pragma unroll
            for (int t = 0; t < 2; ++t) {
                frag16 kf;
                const uint32_t* bp =
                    (const uint32_t*)(K + (size_t)(b * NN + (j + t) * 16 + lrow) * DD + kc * 32)
                    + half * 4;
                kf.q[0] = *(const uint4*)bp;
                kf.q[1] = *(const uint4*)(bp + 8);
                s[t] = __builtin_amdgcn_wmma_f32_16x16x32_bf16(
                    false, qf[kc].v, false, kf.v, (short)0, s[t], false, false);
            }
        }
        if (j + 2 < 64)  // hint next key tile into cache
            __builtin_prefetch(K + (size_t)(b * NN + (j + 2) * 16 + lrow) * DD, 0, 1);

        // ---- scale, leaky relu, mask ----
        #pragma unroll
        for (int t = 0; t < 2; ++t) {
            #pragma unroll
            for (int r = 0; r < 8; ++r) {
                float e = s[t][r] * SCALEV;
                e = e > 0.f ? e : 0.2f * e;
                int g = grow[(size_t)(half * 8 + r) * NN + (j + t) * 16 + lrow];
                s[t][r] = (g == 0) ? NEGV : e;
            }
        }

        // ---- online softmax (row stats across the 16-lane group) ----
        #pragma unroll
        for (int r = 0; r < 8; ++r) {
            float mx = fmaxf(s[0][r], s[1][r]);
            mx = fmaxf(mx, __shfl_xor(mx, 1));
            mx = fmaxf(mx, __shfl_xor(mx, 2));
            mx = fmaxf(mx, __shfl_xor(mx, 4));
            mx = fmaxf(mx, __shfl_xor(mx, 8));
            float mnew  = fmaxf(mrow[r], mx);
            float alpha = __expf(mrow[r] - mnew);
            mrow[r] = mnew;
            float p0 = __expf(s[0][r] - mnew);
            float p1 = __expf(s[1][r] - mnew);
            s[0][r] = p0; s[1][r] = p1;
            float rs = p0 + p1;
            rs += __shfl_xor(rs, 1);
            rs += __shfl_xor(rs, 2);
            rs += __shfl_xor(rs, 4);
            rs += __shfl_xor(rs, 8);
            lsum[r] = lsum[r] * alpha + rs;
            #pragma unroll
            for (int t = 0; t < 8; ++t) o[t][r] *= alpha;
        }

        // ---- stage P (C-layout f32) into LDS as [row][32] bf16, reload as A-frag ----
        #pragma unroll
        for (int t = 0; t < 2; ++t)
            #pragma unroll
            for (int r = 0; r < 8; ++r)
                pl[(half * 8 + r) * 32 + t * 16 + lrow] = (bf16_t)s[t][r];
        // LDS ops are in-order within a wave: reload is safe.
        frag16 pf;
        {
            const uint32_t* ap = (const uint32_t*)(pl + lrow * 32);
            pf.q[0] = *(const uint4*)(ap + half * 4);
            pf.q[1] = *(const uint4*)(ap + 8 + half * 4);
        }

        // ---- O += P @ H (B-frags from transposed h: K contiguous) ----
        const bf16_t* htb = HT + ((size_t)b * DD + dhalf * 128) * NN + j * 32;
        #pragma unroll
        for (int t = 0; t < 8; ++t) {
            frag16 hf;
            const uint32_t* bp = (const uint32_t*)(htb + (size_t)(t * 16 + lrow) * NN) + half * 4;
            hf.q[0] = *(const uint4*)bp;
            hf.q[1] = *(const uint4*)(bp + 8);
            o[t] = __builtin_amdgcn_wmma_f32_16x16x32_bf16(
                false, pf.v, false, hf.v, (short)0, o[t], false, false);
        }
    }

    // ---- epilogue: normalize by l, relu, store f32 + bf16 ----
    #pragma unroll
    for (int r = 0; r < 8; ++r) lsum[r] = 1.f / lsum[r];
    #pragma unroll
    for (int t = 0; t < 8; ++t) {
        const int col = dhalf * 128 + t * 16 + lrow;
        #pragma unroll
        for (int r = 0; r < 8; ++r) {
            float v = o[t][r] * lsum[r];
            v = v > 0.f ? v : 0.f;
            size_t idx = (size_t)(qrow0 + half * 8 + r) * DD + col;
            Of32[idx] = v;
            Obf[idx]  = (bf16_t)v;
        }
    }
}

// ---------------------------------------------------------------------------
extern "C" void kernel_launch(void* const* d_in, const int* in_sizes, int n_in,
                              void* d_out, int out_size, void* d_ws, size_t ws_size,
                              hipStream_t stream) {
    (void)in_sizes; (void)n_in; (void)out_size; (void)ws_size;
    const float* feature = (const float*)d_in[0];
    const int*   graph   = (const int*)d_in[1];
    const float* Wws     = (const float*)d_in[2];
    const float* Wbs     = (const float*)d_in[3];
    const float* Qws     = (const float*)d_in[4];
    const float* Qbs     = (const float*)d_in[5];
    const float* Kws     = (const float*)d_in[6];
    const float* Kbs     = (const float*)d_in[7];
    float* out = (float*)d_out;

    const size_t MN = (size_t)BATCH * NN * DD;   // 4M elements
    const size_t WW = (size_t)2 * DD * DD;       // 131072 elements
    bf16_t* xb  = (bf16_t*)d_ws;
    bf16_t* hb  = xb  + MN;
    bf16_t* hTb = hb  + MN;
    bf16_t* qb  = hTb + MN;
    bf16_t* kb  = qb  + MN;
    bf16_t* wwb = kb  + MN;
    bf16_t* qwb = wwb + WW;
    bf16_t* kwb = qwb + WW;

    // convert inputs/weights to bf16
    cvt_f32_to_bf16<<<(int)(MN / 256), 256, 0, stream>>>(feature, xb, (int)MN);
    cvt_f32_to_bf16<<<(int)(WW / 256), 256, 0, stream>>>(Wws, wwb, (int)WW);
    cvt_f32_to_bf16<<<(int)(WW / 256), 256, 0, stream>>>(Qws, qwb, (int)WW);
    cvt_f32_to_bf16<<<(int)(WW / 256), 256, 0, stream>>>(Kws, kwb, (int)WW);

    dim3 ggrid(128, 2), gblk(256);
    for (int l = 0; l < 2; ++l) {
        const size_t wo = (size_t)l * DD * DD;
        const size_t bo = (size_t)l * DD;
        gemm_bias_bf16<<<ggrid, gblk, 0, stream>>>(xb, wwb + wo, Wbs + bo, hb, hTb);
        gemm_bias_bf16<<<ggrid, gblk, 0, stream>>>(hb, qwb + wo, Qbs + bo, qb, nullptr);
        gemm_bias_bf16<<<ggrid, gblk, 0, stream>>>(hb, kwb + wo, Kbs + bo, kb, nullptr);
        gat_attention<<<256, 256, 0, stream>>>(qb, kb, hTb, graph, out, xb);
    }
}
